// GraphUNet_10694468567510
// MI455X (gfx1250) — matrix operand order repaired and checked by reading, not compiled, not run on hardware
//
#include <hip/hip_runtime.h>
#include <hip/hip_bf16.h>

typedef __attribute__((ext_vector_type(16))) _Float16 v16h;
typedef __attribute__((ext_vector_type(8)))  float    v8f;

#define BM 128
#define BN 64
#define BK 32
#define LSTR 40   // LDS row stride in halves: 80B = 16B aligned, bank-offset 20/row

// Epilogue variants (compile-time):
//  0: plain            C = A@B
//  1: GCN              C = dis[m]*(acc + fix[m]*Ms[m,n]) + bias[n]
//  2: GCN + ReLU
//  3: augment          loads substitute diag:=1, store zeroes diag
// ---------------------------------------------------------------------------
template <int EPI>
__global__ __launch_bounds__(256) void wgemm(
    const float* __restrict__ A, const float* __restrict__ B, float* __restrict__ C,
    int M, int N, int K, int lda, int ldb, int ldc,
    const float* __restrict__ rowscale,   // dis
    const float* __restrict__ rowaddmul,  // fix
    const float* __restrict__ addmat,     // Ms
    int ldadd,
    const float* __restrict__ bias)
{
  (void)M; (void)N; (void)rowscale; (void)rowaddmul; (void)addmat; (void)ldadd; (void)bias;
  __shared__ __align__(16) _Float16 As [2][BM * LSTR];  // [buf][row][k]
  __shared__ __align__(16) _Float16 BsT[2][BN * LSTR];  // [buf][col][k] (transposed)

  const int tid  = threadIdx.x;
  const int lane = tid & 31;
  const int wave = tid >> 5;        // 0..7
  const int wr   = wave >> 1;       // 0..3 : 32-row strip
  const int wc   = wave & 1;        // 0..1 : 32-col strip
  const int lr   = lane & 15;
  const int hs   = lane >> 4;       // half-select
  const int row0 = blockIdx.y * BM;
  const int col0 = blockIdx.x * BN;

  // Per-thread staging coordinates (fixed across K).
  // A tile 128x32: 4 float4 chunks/thread.  B tile 32x64: 2 float4 chunks/thread.
  int aR[4], aC[4], bR[2], bC[2];
  #pragma unroll
  for (int i = 0; i < 4; ++i) { int q = tid + i * 256; aR[i] = q >> 3; aC[i] = (q & 7) * 4; }
  #pragma unroll
  for (int i = 0; i < 2; ++i) { int q = tid + i * 256; bR[i] = q >> 4; bC[i] = (q & 15) * 4; }

  float4 va[4], vb[2];

  auto load_tile = [&](int k0) {
    #pragma unroll
    for (int i = 0; i < 4; ++i) {
      const float* p = A + (size_t)(row0 + aR[i]) * lda + k0 + aC[i];
      va[i] = *reinterpret_cast<const float4*>(p);
      __builtin_prefetch(p + BK, 0, 1);          // speculative: safe past end
    }
    #pragma unroll
    for (int i = 0; i < 2; ++i) {
      const float* p = B + (size_t)(k0 + bR[i]) * ldb + col0 + bC[i];
      vb[i] = *reinterpret_cast<const float4*>(p);
      __builtin_prefetch(p + (size_t)BK * ldb, 0, 1);
    }
  };

  auto store_tile = [&](int buf, int k0) {
    #pragma unroll
    for (int i = 0; i < 4; ++i) {
      float4 v = va[i];
      if constexpr (EPI == 3) {
        int gr = row0 + aR[i], gc = k0 + aC[i];
        if (gr == gc + 0) v.x = 1.0f;
        if (gr == gc + 1) v.y = 1.0f;
        if (gr == gc + 2) v.z = 1.0f;
        if (gr == gc + 3) v.w = 1.0f;
      }
      _Float16* dst = &As[buf][aR[i] * LSTR + aC[i]];
      dst[0] = (_Float16)v.x; dst[1] = (_Float16)v.y;
      dst[2] = (_Float16)v.z; dst[3] = (_Float16)v.w;
    }
    #pragma unroll
    for (int i = 0; i < 2; ++i) {
      float4 v = vb[i];
      if constexpr (EPI == 3) {
        int gk = k0 + bR[i], gc = col0 + bC[i];
        if (gk == gc + 0) v.x = 1.0f;
        if (gk == gc + 1) v.y = 1.0f;
        if (gk == gc + 2) v.z = 1.0f;
        if (gk == gc + 3) v.w = 1.0f;
      }
      _Float16* dst = &BsT[buf][bC[i] * LSTR + bR[i]];
      dst[0 * LSTR] = (_Float16)v.x;
      dst[1 * LSTR] = (_Float16)v.y;
      dst[2 * LSTR] = (_Float16)v.z;
      dst[3 * LSTR] = (_Float16)v.w;
    }
  };

  v8f acc[2][2] = {};

  // Prologue: stage tile 0 into buffer 0.
  load_tile(0);
  store_tile(0, 0);
  __syncthreads();

  int cur = 0;
  for (int k0 = 0; k0 < K; k0 += BK) {
    const int kn = k0 + BK;
    const bool has_next = kn < K;

    // Issue next tile's global loads early; they overlap the MMA phase.
    if (has_next) load_tile(kn);

    // ---- fragments (ISA wave32 16-bit layouts), wide contiguous LDS reads ----
    v16h a[2], b[2];
    #pragma unroll
    for (int i = 0; i < 2; ++i) {
      const _Float16* ap = &As[cur][(wr * 32 + i * 16 + lr) * LSTR + hs * 8];
      #pragma unroll
      for (int t = 0; t < 8; ++t) { a[i][t] = ap[t]; a[i][t + 8] = ap[16 + t]; }
    }
    #pragma unroll
    for (int j = 0; j < 2; ++j) {
      const _Float16* bp = &BsT[cur][(wc * 32 + j * 16 + lr) * LSTR + hs * 8];
      #pragma unroll
      for (int t = 0; t < 8; ++t) { b[j][t] = bp[t]; b[j][t + 8] = bp[16 + t]; }
    }
    #pragma unroll
    for (int i = 0; i < 2; ++i)
      #pragma unroll
      for (int j = 0; j < 2; ++j)
        acc[i][j] = __builtin_amdgcn_wmma_f32_16x16x32_f16(
            false, a[i], false, b[j], (short)0, acc[i][j], false, false);

    // Stage next tile into the other buffer; single barrier per iteration.
    if (has_next) {
      store_tile(cur ^ 1, kn);
      __syncthreads();
      cur ^= 1;
    }
  }

  // ---- epilogue: VGPR r -> M = r + 8*hs, N = lr within each 16x16 tile ----
  #pragma unroll
  for (int i = 0; i < 2; ++i) {
    #pragma unroll
    for (int j = 0; j < 2; ++j) {
      const int mb = row0 + wr * 32 + i * 16 + hs * 8;
      const int nn = col0 + wc * 32 + j * 16 + lr;
      #pragma unroll
      for (int r = 0; r < 8; ++r) {
        int m = mb + r;
        float v = acc[i][j][r];
        if constexpr (EPI == 1 || EPI == 2) {
          v += rowaddmul[m] * addmat[(size_t)m * ldadd + nn];
          v *= rowscale[m];
          v += bias[nn];
          if constexpr (EPI == 2) v = v > 0.f ? v : 0.f;
        }
        if constexpr (EPI == 3) {
          if (m == nn) v = 0.f;
        }
        C[(size_t)m * ldc + nn] = v;
      }
    }
  }
}

// ---------------------------------------------------------------------------
// Small helper kernels
// ---------------------------------------------------------------------------
__global__ void build_adj(const int* __restrict__ ei, float* __restrict__ A,
                          int E, int n) {
  int e = blockIdx.x * blockDim.x + threadIdx.x;
  if (e >= E) return;
  int src = ei[e];
  int dst = ei[E + e];
  atomicAdd(&A[(size_t)dst * n + src], 1.0f);   // integer adds: order-exact
}

__global__ void row_stats(const float* __restrict__ A, int n,
                          float* __restrict__ dis, float* __restrict__ fix) {
  int row = blockIdx.x;
  const float* Ar = A + (size_t)row * n;
  float s = 0.f;
  for (int j = threadIdx.x; j < n; j += blockDim.x) s += Ar[j];
  __shared__ float red[256];
  red[threadIdx.x] = s;
  __syncthreads();
  for (int off = 128; off > 0; off >>= 1) {
    if (threadIdx.x < off) red[threadIdx.x] += red[threadIdx.x + off];
    __syncthreads();
  }
  if (threadIdx.x == 0) {
    float f = (Ar[row] == 0.0f) ? 1.0f : 0.0f;   // self-loop only if diag empty
    float d = red[0] + f;
    dis[row] = d > 0.f ? rsqrtf(d) : 0.0f;
    fix[row] = f;
  }
}

__global__ void scale_rows(const float* __restrict__ M, const float* __restrict__ dis,
                           float* __restrict__ Ms, int total) {
  int idx = blockIdx.x * blockDim.x + threadIdx.x;
  if (idx >= total) return;
  Ms[idx] = M[idx] * dis[idx >> 7];              // 128 cols
}

__global__ void score_kernel(const float* __restrict__ X, const float* __restrict__ w,
                             float* __restrict__ s, int n) {
  int i = blockIdx.x * blockDim.x + threadIdx.x;
  if (i >= n) return;
  float nrm = 0.f;
  for (int j = 0; j < 128; ++j) { float v = w[j]; nrm += v * v; }
  nrm = sqrtf(nrm);
  const float* xr = X + (size_t)i * 128;
  float d = 0.f;
  for (int j = 0; j < 128; ++j) d += xr[j] * w[j];
  s[i] = tanhf(d / nrm);
}

// Single-block bitonic sort (descending score, ascending index tie-break).
// n is a power of two <= 4096.
__global__ void topk_sort(const float* __restrict__ score, int n, int k,
                          float* __restrict__ vals, int* __restrict__ perm) {
  __shared__ float s[4096];
  __shared__ int  id[4096];
  for (int i = threadIdx.x; i < n; i += blockDim.x) { s[i] = score[i]; id[i] = i; }
  __syncthreads();
  for (int size = 2; size <= n; size <<= 1) {
    for (int stride = size >> 1; stride > 0; stride >>= 1) {
      for (int i = threadIdx.x; i < n; i += blockDim.x) {
        int j = i ^ stride;
        if (j > i) {
          float si = s[i], sj = s[j];
          int   ii = id[i], ij = id[j];
          bool cmp = (si > sj) || (si == sj && ii < ij);  // i-elem first?
          bool dir = ((i & size) == 0);
          if (cmp != dir) { s[i] = sj; s[j] = si; id[i] = ij; id[j] = ii; }
        }
      }
      __syncthreads();
    }
  }
  for (int i = threadIdx.x; i < k; i += blockDim.x) { vals[i] = s[i]; perm[i] = id[i]; }
}

__global__ void gather_x(const float* __restrict__ X, const int* __restrict__ perm,
                         const float* __restrict__ vals, float* __restrict__ Xo, int k) {
  int idx = blockIdx.x * blockDim.x + threadIdx.x;
  if (idx >= k * 128) return;
  int i = idx >> 7, j = idx & 127;
  Xo[idx] = X[(size_t)perm[i] * 128 + j] * vals[i];
}

__global__ void gather_A(const float* __restrict__ A, const int* __restrict__ perm,
                         float* __restrict__ Ao, int nold, int k) {
  int idx = blockIdx.x * blockDim.x + threadIdx.x;
  if (idx >= k * k) return;
  int i = idx / k, j = idx % k;
  Ao[(size_t)i * k + j] = A[(size_t)perm[i] * nold + perm[j]];
}

__global__ void mean_pool(const float* __restrict__ X, int n, float* __restrict__ ge) {
  int j = threadIdx.x;            // 128 threads
  float s = 0.f;
  for (int i = 0; i < n; ++i) s += X[(size_t)i * 128 + j];
  ge[j] = s / (float)n;
}

__global__ void final_linear(const float* __restrict__ ge, const float* __restrict__ Wl,
                             const float* __restrict__ bl, float* __restrict__ out) {
  int o = threadIdx.x;
  if (o >= 16) return;
  float s = bl[o];
  for (int j = 0; j < 128; ++j) s += ge[j] * Wl[j * 16 + o];
  out[o] = s;
}

// ---------------------------------------------------------------------------
// Host-side orchestration
// ---------------------------------------------------------------------------
static void gcn(hipStream_t stream, const float* A, int n, const float* Xin, int cin,
                const float* W, const float* bias, bool relu, float* Xout,
                float* dis, float* fix, float* Mb, float* Msb) {
  row_stats<<<n, 256, 0, stream>>>(A, n, dis, fix);
  dim3 g1(128 / BN, n / BM);
  wgemm<0><<<g1, 256, 0, stream>>>(Xin, W, Mb, n, 128, cin, cin, 128, 128,
                                   nullptr, nullptr, nullptr, 0, nullptr);
  int tot = n * 128;
  scale_rows<<<(tot + 255) / 256, 256, 0, stream>>>(Mb, dis, Msb, tot);
  dim3 g2(128 / BN, n / BM);
  if (relu)
    wgemm<2><<<g2, 256, 0, stream>>>(A, Msb, Xout, n, 128, n, n, 128, 128,
                                     dis, fix, Msb, 128, bias);
  else
    wgemm<1><<<g2, 256, 0, stream>>>(A, Msb, Xout, n, 128, n, n, 128, 128,
                                     dis, fix, Msb, 128, bias);
}

extern "C" void kernel_launch(void* const* d_in, const int* in_sizes, int n_in,
                              void* d_out, int out_size, void* d_ws, size_t ws_size,
                              hipStream_t stream) {
  (void)in_sizes; (void)n_in; (void)out_size; (void)ws_size;
  const float* x  = (const float*)d_in[0];
  const int*   ei = (const int*)d_in[1];
  const float* W0 = (const float*)d_in[2];
  const float* b0 = (const float*)d_in[3];
  const float* Wh = (const float*)d_in[4];   // (7,128,128)
  const float* bh = (const float*)d_in[5];   // (7,128)
  const float* pw = (const float*)d_in[6];   // (3,128)
  const float* Wg = (const float*)d_in[7];
  const float* bg = (const float*)d_in[8];
  const float* Wl = (const float*)d_in[9];
  const float* bl = (const float*)d_in[10];
  float* out = (float*)d_out;

  const size_t NN = 4096;
  float* ws    = (float*)d_ws;
  float* Abuf  = ws;                         // 4096^2
  float* Abuf2 = Abuf  + NN * NN;            // 4096^2
  float* Xa    = Abuf2 + NN * NN;            // 4096*128
  float* Xb    = Xa + NN * 128;
  float* Mb    = Xb + NN * 128;
  float* Msb   = Mb + NN * 128;
  float* dis   = Msb + NN * 128;
  float* fix   = dis + NN;
  float* score = fix + NN;
  float* vals  = score + NN;
  int*   perm  = (int*)(vals + NN);
  float* ge    = (float*)(perm + NN);

  // Build dense adjacency: A[dst,src] += 1
  hipMemsetAsync(Abuf, 0, NN * NN * sizeof(float), stream);
  build_adj<<<65536 / 256, 256, 0, stream>>>(ei, Abuf, 65536, 4096);

  float* A  = Abuf;
  float* A2 = Abuf2;
  float* Xc = Xa;
  float* Xn = Xb;

  gcn(stream, A, 4096, x,  64,  W0, b0, false, Xc, dis, fix, Mb, Msb);
  gcn(stream, A, 4096, Xc, 128, Wh, bh, true,  Xn, dis, fix, Mb, Msb);
  { float* t = Xc; Xc = Xn; Xn = t; }

  int n = 4096;
  for (int i = 1; i <= 3; ++i) {
    // augment_adj: A <- zerodiag( (A with diag:=1) @ (A with diag:=1) )
    dim3 ga(n / BN, n / BM);
    wgemm<3><<<ga, 256, 0, stream>>>(A, A, A2, n, n, n, n, n, n,
                                     nullptr, nullptr, nullptr, 0, nullptr);
    { float* t = A; A = A2; A2 = t; }

    int k = n / 2;
    score_kernel<<<(n + 255) / 256, 256, 0, stream>>>(Xc, pw + (size_t)(i - 1) * 128, score, n);
    topk_sort<<<1, 1024, 0, stream>>>(score, n, k, vals, perm);
    gather_x<<<(k * 128 + 255) / 256, 256, 0, stream>>>(Xc, perm, vals, Xn, k);
    gather_A<<<(k * k + 255) / 256, 256, 0, stream>>>(A, perm, A2, n, k);
    { float* t = A; A = A2; A2 = t; }
    { float* t = Xc; Xc = Xn; Xn = t; }
    n = k;

    gcn(stream, A, n, Xc, 128, Wh + (size_t)(2 * i - 1) * 128 * 128,
        bh + (size_t)(2 * i - 1) * 128, false, Xn, dis, fix, Mb, Msb);
    gcn(stream, A, n, Xn, 128, Wh + (size_t)(2 * i) * 128 * 128,
        bh + (size_t)(2 * i) * 128, true, Xc, dis, fix, Mb, Msb);
  }

  // conv_to_gr_embedding + relu (H -> DG, both 128)
  gcn(stream, A, n, Xc, 128, Wg, bg, true, Xn, dis, fix, Mb, Msb);

  mean_pool<<<1, 128, 0, stream>>>(Xn, n, ge);
  final_linear<<<1, 16, 0, stream>>>(ge, Wl, bl, out);   // preds -> out[0:16]

  // Remaining outputs, flattened in reference return order.
  hipMemcpyAsync(out + 16, Xn, (size_t)n * 128 * sizeof(float),
                 hipMemcpyDeviceToDevice, stream);
  hipMemcpyAsync(out + 16 + (size_t)n * 128, A, (size_t)n * n * sizeof(float),
                 hipMemcpyDeviceToDevice, stream);
  hipMemcpyAsync(out + 16 + (size_t)n * 128 + (size_t)n * n, ge, 128 * sizeof(float),
                 hipMemcpyDeviceToDevice, stream);
}
